// FeatureEncoding_438086664760
// MI455X (gfx1250) — matrix operand order repaired and checked by reading, not compiled, not run on hardware
//
#include <hip/hip_runtime.h>

#define BATCH   4
#define NPTS    16384
#define NSAMP   2048
#define KNN     32
#define CIN     64
#define R2      0.16f     // radius^2, radius = 0.4
#define D0      64
#define D1      64
#define D2      128
#define K0PAD   96        // 3+64=67 padded up to 3 K-steps of 32

// packed-fragment element counts (halves): [ksteps][ntiles][32 lanes][16 halves]
#define P0      (3*4*32*16)   // 6144
#define P1      (2*4*32*16)   // 4096
#define P2      (2*8*32*16)   // 8192

typedef _Float16 v16h __attribute__((ext_vector_type(16)));
typedef float    v8f  __attribute__((ext_vector_type(8)));

// ---------------------------------------------------------------------------
// 1) Farthest point sampling: one workgroup per batch, min-dist array in LDS.
//    Writes selected coordinates directly into d_out's new_xyz region.
// ---------------------------------------------------------------------------
__global__ __launch_bounds__(1024) void fps_kernel(const float* __restrict__ xyz,
                                                   float* __restrict__ out_xyz)
{
    extern __shared__ float dist_s[];                 // NPTS floats (64 KB dynamic LDS)
    __shared__ float red_v[1024];
    __shared__ int   red_i[1024];
    __shared__ int   s_last;

    const int b = blockIdx.x;
    const int t = threadIdx.x;
    const float* X = xyz + (size_t)b * NPTS * 3;

    for (int n = t; n < NPTS; n += 1024) dist_s[n] = 1e10f;
    if (t == 0) {
        s_last = 0;
        out_xyz[(size_t)b * NSAMP * 3 + 0] = X[0];
        out_xyz[(size_t)b * NSAMP * 3 + 1] = X[1];
        out_xyz[(size_t)b * NSAMP * 3 + 2] = X[2];
    }
    __syncthreads();

    for (int s = 1; s < NSAMP; ++s) {
        const int last = s_last;
        const float cx = X[last*3+0], cy = X[last*3+1], cz = X[last*3+2];
        float bestv = -1.0f; int besti = 0;
        #pragma unroll
        for (int j = 0; j < NPTS/1024; ++j) {
            const int n = t + j*1024;
            const float dx = X[n*3+0]-cx, dy = X[n*3+1]-cy, dz = X[n*3+2]-cz;
            const float d  = dx*dx + dy*dy + dz*dz;
            const float nd = fminf(dist_s[n], d);
            dist_s[n] = nd;
            if (nd > bestv) { bestv = nd; besti = n; }
        }
        red_v[t] = bestv; red_i[t] = besti;
        __syncthreads();
        for (int off = 512; off > 0; off >>= 1) {
            if (t < off) {
                const float v2 = red_v[t+off]; const int i2 = red_i[t+off];
                if (v2 > red_v[t] || (v2 == red_v[t] && i2 < red_i[t])) {
                    red_v[t] = v2; red_i[t] = i2;    // argmax, first-index tie-break
                }
            }
            __syncthreads();
        }
        if (t == 0) {
            const int nxt = red_i[0];
            s_last = nxt;
            out_xyz[((size_t)b*NSAMP + s)*3 + 0] = X[nxt*3+0];
            out_xyz[((size_t)b*NSAMP + s)*3 + 1] = X[nxt*3+1];
            out_xyz[((size_t)b*NSAMP + s)*3 + 2] = X[nxt*3+2];
        }
        __syncthreads();
    }
}

// ---------------------------------------------------------------------------
// 2) Ball query: one workgroup (256 thr) per query point. d2 row lives in LDS;
//    32 incremental argmin extractions; only the winning 64-entry chunk rescans.
// ---------------------------------------------------------------------------
__global__ __launch_bounds__(256) void ball_query_kernel(const float* __restrict__ xyz,
                                                         const float* __restrict__ new_xyz,
                                                         int* __restrict__ grp_idx)
{
    extern __shared__ float d2_s[];                   // NPTS floats (64 KB dynamic LDS)
    __shared__ float sv[256]; __shared__ int si[256]; // cached per-chunk minima
    __shared__ float rv[256]; __shared__ int ri[256]; // reduction scratch

    const int t  = threadIdx.x;
    const int bs = blockIdx.x;
    const int b  = bs / NSAMP;
    const float* X = xyz + (size_t)b * NPTS * 3;

    const float qx = new_xyz[(size_t)bs*3+0];
    const float qy = new_xyz[(size_t)bs*3+1];
    const float qz = new_xyz[(size_t)bs*3+2];

    const int base = t * (NPTS/256);                  // contiguous chunk of 64
    float mv = 3.4e38f; int mi = base;
    #pragma unroll 4
    for (int j = 0; j < NPTS/256; ++j) {
        const int n = base + j;
        const float dx = X[n*3+0]-qx, dy = X[n*3+1]-qy, dz = X[n*3+2]-qz;
        const float d  = dx*dx + dy*dy + dz*dz;
        d2_s[n] = d;
        if (d < mv) { mv = d; mi = n; }               // strict < keeps smallest index
    }
    sv[t] = mv; si[t] = mi;
    __syncthreads();

    int nearest = 0;
    for (int k = 0; k < KNN; ++k) {
        rv[t] = sv[t]; ri[t] = si[t];
        __syncthreads();
        for (int off = 128; off > 0; off >>= 1) {
            if (t < off) {
                const float v2 = rv[t+off]; const int i2 = ri[t+off];
                if (v2 < rv[t] || (v2 == rv[t] && i2 < ri[t])) { rv[t] = v2; ri[t] = i2; }
            }
            __syncthreads();
        }
        const float bv = rv[0]; const int bi = ri[0];
        if (k == 0) nearest = bi;
        if (t == 0) grp_idx[(size_t)bs*KNN + k] = (bv <= R2) ? bi : nearest;
        if (t == (bi >> 6)) {                          // owner chunk rescans
            d2_s[bi] = 3.4e38f;
            float nv = 3.4e38f; int ni = base;
            for (int j = 0; j < NPTS/256; ++j) {
                const float d = d2_s[base + j];
                if (d < nv) { nv = d; ni = base + j; }
            }
            sv[t] = nv; si[t] = ni;
        }
        __syncthreads();
    }
}

// ---------------------------------------------------------------------------
// 3) Weight conversion + pre-swizzle into WMMA B-fragment order.
//    Fragment (ks,nt): lane holds column n = nt*16+(lane&15), rows
//    kb = ks*32+(j>>3)*16+(lane>>4)*8+(j&7) (j=0..15) -> 16 contiguous halves per lane,
//    so the MLP kernel reads each B operand as one 32-byte per-lane load
//    (2x global_load_b128, fully coalesced, L2-resident).
// ---------------------------------------------------------------------------
__global__ __launch_bounds__(256) void cvt_pack_weights(const float* __restrict__ W0,
                                                        const float* __restrict__ W1,
                                                        const float* __restrict__ W2,
                                                        _Float16* __restrict__ W0p,
                                                        _Float16* __restrict__ W1p,
                                                        _Float16* __restrict__ W2p)
{
    int i = blockIdx.x * 256 + threadIdx.x;
    if (i < P0) {                                      // layer 0: [96][64], zero-pad K>=67
        const int j = i & 15, lane = (i >> 4) & 31, rest = i >> 9;
        const int nt = rest & 3, ks = rest >> 2;
        const int n  = nt*16 + (lane & 15);
        const int kb = ks*32 + (j >> 3)*16 + (lane >> 4)*8 + (j & 7);
        W0p[i] = (_Float16)(kb < 3+CIN ? W0[kb*D0 + n] : 0.0f);
        return;
    }
    i -= P0;
    if (i < P1) {                                      // layer 1: [64][64]
        const int j = i & 15, lane = (i >> 4) & 31, rest = i >> 9;
        const int nt = rest & 3, ks = rest >> 2;
        const int n  = nt*16 + (lane & 15);
        const int kb = ks*32 + (j >> 3)*16 + (lane >> 4)*8 + (j & 7);
        W1p[i] = (_Float16)W1[kb*D1 + n];
        return;
    }
    i -= P1;
    if (i < P2) {                                      // layer 2: [64][128]
        const int j = i & 15, lane = (i >> 4) & 31, rest = i >> 9;
        const int nt = rest & 7, ks = rest >> 3;
        const int n  = nt*16 + (lane & 15);
        const int kb = ks*32 + (j >> 3)*16 + (lane >> 4)*8 + (j & 7);
        W2p[i] = (_Float16)W2[kb*D2 + n];
    }
}

// ---------------------------------------------------------------------------
// 4) Gather + 3-layer MLP via v_wmma_f32_16x16x32_f16 + max over K.
//    One workgroup (8 waves) per query point; each wave owns one 16x16 tile.
//    A fragment: lane m = lane&15, ko = (lane>>4)*8 -> halves [ko..ko+7] and
//    [16+ko..16+ko+7] (two contiguous 16B LDS runs, per ISA 16-bit A layout).
//    B fragment: one contiguous 32B per-lane load from the packed weights.
//    C/D: VGPR r holds (M=r | M=8+r) x (N = lane&15); bias is seeded into C.
// ---------------------------------------------------------------------------
__global__ __launch_bounds__(256) void group_mlp_kernel(
    const float* __restrict__ xyz, const float* __restrict__ feat,
    const float* __restrict__ new_xyz, const int* __restrict__ grp_idx,
    const _Float16* __restrict__ W0p, const float* __restrict__ b0,
    const _Float16* __restrict__ W1p, const float* __restrict__ b1,
    const _Float16* __restrict__ W2p, const float* __restrict__ b2,
    float* __restrict__ out_feat)
{
    __shared__ __align__(16) _Float16 g [KNN][K0PAD];
    __shared__ __align__(16) _Float16 h0[KNN][D0];
    __shared__ __align__(16) _Float16 h1[KNN][D1];
    __shared__ __align__(16) float    h2[KNN][D2];
    __shared__ int   idx_s[KNN];
    __shared__ float q[3];

    const int t  = threadIdx.x;
    const int bs = blockIdx.x;
    const int b  = bs / NSAMP;

    __builtin_prefetch(W0p, 0, 0);
    __builtin_prefetch(W1p, 0, 0);
    __builtin_prefetch(W2p, 0, 0);

    if (t < KNN) idx_s[t] = grp_idx[(size_t)bs*KNN + t];
    if (t < 3)   q[t] = new_xyz[(size_t)bs*3 + t];
    __syncthreads();

    // Build g = [rel_xyz(3) ; features(64) ; zero-pad(29)] in f16.
    for (int e = t; e < KNN*K0PAD; e += 256) {
        const int r = e / K0PAD, c = e % K0PAD;
        const int n = idx_s[r];
        float v = 0.0f;
        if (c < 3)          v = xyz [((size_t)b*NPTS + n)*3   + c] - q[c];
        else if (c < 3+CIN) v = feat[((size_t)b*NPTS + n)*CIN + (c-3)];
        g[r][c] = (_Float16)v;
    }
    __syncthreads();

    const int lane = t & 31;
    const int w    = t >> 5;           // wave id 0..7
    const int lm   = lane & 15;
    const int hi   = lane >> 4;
    const int mt   = w >> 2;           // M tile 0..1 (rows of 16 neighbors)
    const int M    = mt*16 + lm;

    // ---- Layer 0: [32x96] x [96x64] ----
    {
        const int nt = w & 3;
        const float bias = b0[nt*16 + lm];
        v8f acc = {bias,bias,bias,bias,bias,bias,bias,bias};
        #pragma unroll
        for (int ks = 0; ks < 3; ++ks) {
            v16h a;
            const _Float16* pa = &g[M][ks*32 + hi*8];
            #pragma unroll
            for (int i = 0; i < 8; ++i) { a[i] = pa[i]; a[8+i] = pa[16+i]; }
            const v16h bm = *(const v16h*)(W0p + (((ks*4) + nt)*32 + lane)*16);
            acc = __builtin_amdgcn_wmma_f32_16x16x32_f16(false, a, false, bm,
                                                         (short)0, acc, false, false);
        }
        #pragma unroll
        for (int r = 0; r < 8; ++r)
            h0[mt*16 + hi*8 + r][nt*16 + lm] = (_Float16)fmaxf(acc[r], 0.0f);
    }
    __syncthreads();

    // ---- Layer 1: [32x64] x [64x64] ----
    {
        const int nt = w & 3;
        const float bias = b1[nt*16 + lm];
        v8f acc = {bias,bias,bias,bias,bias,bias,bias,bias};
        #pragma unroll
        for (int ks = 0; ks < 2; ++ks) {
            v16h a;
            const _Float16* pa = &h0[M][ks*32 + hi*8];
            #pragma unroll
            for (int i = 0; i < 8; ++i) { a[i] = pa[i]; a[8+i] = pa[16+i]; }
            const v16h bm = *(const v16h*)(W1p + (((ks*4) + nt)*32 + lane)*16);
            acc = __builtin_amdgcn_wmma_f32_16x16x32_f16(false, a, false, bm,
                                                         (short)0, acc, false, false);
        }
        #pragma unroll
        for (int r = 0; r < 8; ++r)
            h1[mt*16 + hi*8 + r][nt*16 + lm] = (_Float16)fmaxf(acc[r], 0.0f);
    }
    __syncthreads();

    // ---- Layer 2: [32x64] x [64x128]; each wave does 2 N-tiles ----
    #pragma unroll
    for (int half = 0; half < 2; ++half) {
        const int nt = (w & 3)*2 + half;
        const float bias = b2[nt*16 + lm];
        v8f acc = {bias,bias,bias,bias,bias,bias,bias,bias};
        #pragma unroll
        for (int ks = 0; ks < 2; ++ks) {
            v16h a;
            const _Float16* pa = &h1[M][ks*32 + hi*8];
            #pragma unroll
            for (int i = 0; i < 8; ++i) { a[i] = pa[i]; a[8+i] = pa[16+i]; }
            const v16h bm = *(const v16h*)(W2p + (((ks*8) + nt)*32 + lane)*16);
            acc = __builtin_amdgcn_wmma_f32_16x16x32_f16(false, a, false, bm,
                                                         (short)0, acc, false, false);
        }
        #pragma unroll
        for (int r = 0; r < 8; ++r)
            h2[mt*16 + hi*8 + r][nt*16 + lm] = fmaxf(acc[r], 0.0f);
    }
    __syncthreads();

    // ---- Max over K=32 neighbors ----
    if (t < D2) {
        float m = h2[0][t];
        #pragma unroll 8
        for (int r = 1; r < KNN; ++r) m = fmaxf(m, h2[r][t]);
        out_feat[(size_t)bs*D2 + t] = m;
    }
}

// ---------------------------------------------------------------------------
extern "C" void kernel_launch(void* const* d_in, const int* in_sizes, int n_in,
                              void* d_out, int out_size, void* d_ws, size_t ws_size,
                              hipStream_t stream)
{
    (void)in_sizes; (void)n_in; (void)out_size; (void)ws_size;

    const float* xyz  = (const float*)d_in[0];
    const float* feat = (const float*)d_in[1];
    const float* W0   = (const float*)d_in[2];
    const float* b0   = (const float*)d_in[3];
    const float* W1   = (const float*)d_in[4];
    const float* b1   = (const float*)d_in[5];
    const float* W2   = (const float*)d_in[6];
    const float* b2   = (const float*)d_in[7];

    float* out_xyz  = (float*)d_out;                         // [B,S,3]
    float* out_feat = out_xyz + (size_t)BATCH*NSAMP*3;       // [B,S,128]

    char* p = (char*)d_ws;
    int*      grp_idx = (int*)p;      p += (size_t)BATCH*NSAMP*KNN*sizeof(int);
    _Float16* W0p     = (_Float16*)p; p += (size_t)P0*sizeof(_Float16);
    _Float16* W1p     = (_Float16*)p; p += (size_t)P1*sizeof(_Float16);
    _Float16* W2p     = (_Float16*)p; p += (size_t)P2*sizeof(_Float16);

    fps_kernel<<<BATCH, 1024, NPTS*sizeof(float), stream>>>(xyz, out_xyz);

    cvt_pack_weights<<<(P0 + P1 + P2 + 255)/256, 256, 0, stream>>>(
        W0, W1, W2, W0p, W1p, W2p);

    ball_query_kernel<<<BATCH*NSAMP, 256, NPTS*sizeof(float), stream>>>(
        xyz, out_xyz, grp_idx);

    group_mlp_kernel<<<BATCH*NSAMP, 256, 0, stream>>>(
        xyz, feat, out_xyz, grp_idx, W0p, b0, W1p, b1, W2p, b2, out_feat);
}
